// PoolExpandTransformerModel_7404523618742
// MI455X (gfx1250) — compile-verified, matrix operand order checked by reading, not run on hardware
//
#include <hip/hip_runtime.h>
#include <hip/hip_bf16.h>

// ---------------- model constants ----------------
#define B_      2
#define S_PRE   2048
#define D_      1024
#define H_      16
#define FF_     4096
#define NT_     32000
#define CF_     4
#define DH_     64
#define S_POST  512          // S_PRE / CF_
#define EPS_    1e-5f

#define HAVE_TDM 1           // tensor_load_to_lds builtin is declared on both toolchains

// ---------------- types ----------------
typedef __attribute__((ext_vector_type(16))) __bf16        v16bf;
typedef __attribute__((ext_vector_type(8)))  float         v8f;
typedef __attribute__((ext_vector_type(4)))  unsigned int  u32x4;
typedef __attribute__((ext_vector_type(4)))  int           i32x4;
typedef __attribute__((ext_vector_type(8)))  int           i32x8;

union FragU { u32x4 u[2]; v16bf v; };

__device__ __forceinline__ unsigned short f2bf(float x) {
  unsigned int u = __float_as_uint(x);
  u += 0x7FFFu + ((u >> 16) & 1u);      // round-to-nearest-even
  return (unsigned short)(u >> 16);
}

// WMMA 16x16x32 bf16 -> f32 accumulate
#define WMMA_BF16(Af, Bf, Cf_) \
  __builtin_amdgcn_wmma_f32_16x16x32_bf16(false, (Af), false, (Bf), (short)0, (Cf_), false, false)

// Load one 16x32 bf16 fragment from a row-major slab with row stride ldk.
// lane<16 : row=lane,    K chunks {0..7} and {16..23}
// lane>=16: row=lane-16, K chunks {8..15} and {24..31}
__device__ __forceinline__ v16bf load_frag_g(const unsigned short* p, int ldk, int lane) {
  const int r  = lane & 15;
  const int hi = lane >> 4;
  const unsigned short* q = p + (long long)r * ldk + hi * 8;
  FragU f;
  f.u[0] = *(const u32x4*)(q);
  f.u[1] = *(const u32x4*)(q + 16);
  return f.v;
}

// Same fragment pattern out of an LDS tile with padded row stride of 40 elements
// (32 data + 8 pad -> 80B row stride, conflict-free across the 16-lane groups).
#define LDS_ROW_STRIDE 40
__device__ __forceinline__ v16bf load_frag_lds(const unsigned short* t, int row0, int lane) {
  const int r  = lane & 15;
  const int hi = lane >> 4;
  const unsigned short* q = t + (row0 + r) * LDS_ROW_STRIDE + hi * 8;
  FragU f;
  f.u[0] = *(const u32x4*)(q);
  f.u[1] = *(const u32x4*)(q + 16);
  return f.v;
}

#ifdef HAVE_TDM
// ---------------- Tensor Data Mover: 2D tile load, 128 rows x 32 cols bf16 ----------
// Builds the D# per CDNA5 ISA 8.3/8.4 and issues TENSOR_LOAD_TO_LDS.
// Pads LDS destination: after every 16 DWORDs (one 64B row) insert 4 DWORDs (16B)
// -> LDS row stride 80B = LDS_ROW_STRIDE elements.
__device__ __forceinline__ void tdm_load_tile_128x32(
    unsigned lds_byte_off, const unsigned short* gsrc,
    unsigned K_elems, unsigned rows_total)
{
  const unsigned long long ga = (unsigned long long)gsrc;
  u32x4 g0 = { 1u,                                    // count=1, user mode, no gather
               lds_byte_off,                          // lds_addr (bytes)
               (unsigned)(ga & 0xFFFFFFFFull),        // global_addr[31:0]
               (unsigned)((ga >> 32) & 0x01FFFFFFull) | (2u << 30) };  // addr[56:32] | type=2
  const unsigned flags = 0x00010000u                  // data_size = 1 -> 2 bytes
                       | 0x00100000u                  // pad_enable
                       | (3u << 22)                   // pad_interval: 16 DWORDs
                       | (3u << 25);                  // pad_amount:   4 DWORDs
  // group1 bit layout (ISA 8.4), assembled as 4x64b then split to dwords
  const unsigned long long q0 = (unsigned long long)flags
                              | ((unsigned long long)(K_elems & 0xFFFFu) << 48);    // tensor_dim0 lo
  const unsigned long long q1 = (unsigned long long)(K_elems >> 16)                 // tensor_dim0 hi
                              | ((unsigned long long)rows_total << 16)              // tensor_dim1
                              | (32ull << 48);                                      // tile_dim0 = 32
  const unsigned long long q2 = 128ull                                              // tile_dim1 = 128
                              | ((unsigned long long)K_elems << 32);                // dim0_stride lo32
  const unsigned long long q3 = 0ull;                                               // stride hi, dim1_stride=0
  i32x8 g1 = { (int)q0, (int)(q0 >> 32), (int)q1, (int)(q1 >> 32),
               (int)q2, (int)(q2 >> 32), (int)q3, (int)(q3 >> 32) };
  i32x4 gz = { 0, 0, 0, 0 };
#if defined(__clang_major__) && __clang_major__ >= 23
  i32x8 z8 = { 0, 0, 0, 0, 0, 0, 0, 0 };
  __builtin_amdgcn_tensor_load_to_lds(g0, g1, gz, gz, z8, 0);
#else
  __builtin_amdgcn_tensor_load_to_lds(g0, g1, gz, gz, 0);
#endif
}

// ---------------- TDM-staged WMMA GEMM: C[M,N] = A[M,K] @ W[N,K]^T (+bias)(+relu) ---
// block = 256 threads (8 waves), block tile = 128(M) x 128(N).
// Per k-step (32): TDM pulls A-tile(128x32) and B-tile(128x32) into padded LDS,
// double-buffered; each wave computes a 32x64 tile = 8 WMMAs from ds_load frags.
// M%128==0, N%128==0, K%32==0.
__global__ __launch_bounds__(256) void gemm_bf16_tdm_kernel(
    const unsigned short* __restrict__ A,
    const unsigned short* __restrict__ W,
    const float*          __restrict__ bias,   // nullptr ok
    float*                __restrict__ Cf,     // f32 out (or nullptr)
    unsigned short*       __restrict__ Cb,     // bf16 out (or nullptr)
    int M, int N, int K, int relu)
{
  __shared__ __align__(16) unsigned short ldsA[2][128 * LDS_ROW_STRIDE];
  __shared__ __align__(16) unsigned short ldsB[2][128 * LDS_ROW_STRIDE];

  const int lane = threadIdx.x & 31;
  const int wave = threadIdx.x >> 5;
  const int wm   = wave >> 1;                      // 0..3 : 32 rows each
  const int wn   = wave & 1;                       // 0..1 : 64 cols each
  const long long mBase = (long long)blockIdx.y * 128;
  const long long nBase = (long long)blockIdx.x * 128;

  const unsigned short* Ag = A + mBase * K;
  const unsigned short* Wg = W + nBase * K;

  if (wave == 0) {
    tdm_load_tile_128x32((unsigned)(unsigned long long)&ldsA[0][0], Ag, (unsigned)K, (unsigned)M);
    tdm_load_tile_128x32((unsigned)(unsigned long long)&ldsB[0][0], Wg, (unsigned)K, (unsigned)N);
    __builtin_amdgcn_s_wait_tensorcnt(0);
  }
  __syncthreads();

  v8f acc[2][4];
  #pragma unroll
  for (int i = 0; i < 2; ++i)
    for (int j = 0; j < 4; ++j)
      for (int r = 0; r < 8; ++r) acc[i][j][r] = 0.f;

  const int nSteps = K >> 5;
  for (int s = 0; s < nSteps; ++s) {
    const int cur = s & 1;
    if (s + 1 < nSteps && wave == 0) {
      const int k1 = (s + 1) << 5;
      tdm_load_tile_128x32((unsigned)(unsigned long long)&ldsA[cur ^ 1][0], Ag + k1, (unsigned)K, (unsigned)M);
      tdm_load_tile_128x32((unsigned)(unsigned long long)&ldsB[cur ^ 1][0], Wg + k1, (unsigned)K, (unsigned)N);
    }
    const unsigned short* At = ldsA[cur];
    const unsigned short* Bt = ldsB[cur];
    v16bf a0 = load_frag_lds(At, wm * 32,      lane);
    v16bf a1 = load_frag_lds(At, wm * 32 + 16, lane);
    v16bf b0 = load_frag_lds(Bt, wn * 64,      lane);
    v16bf b1 = load_frag_lds(Bt, wn * 64 + 16, lane);
    v16bf b2 = load_frag_lds(Bt, wn * 64 + 32, lane);
    v16bf b3 = load_frag_lds(Bt, wn * 64 + 48, lane);
    acc[0][0] = WMMA_BF16(a0, b0, acc[0][0]);
    acc[0][1] = WMMA_BF16(a0, b1, acc[0][1]);
    acc[0][2] = WMMA_BF16(a0, b2, acc[0][2]);
    acc[0][3] = WMMA_BF16(a0, b3, acc[0][3]);
    acc[1][0] = WMMA_BF16(a1, b0, acc[1][0]);
    acc[1][1] = WMMA_BF16(a1, b1, acc[1][1]);
    acc[1][2] = WMMA_BF16(a1, b2, acc[1][2]);
    acc[1][3] = WMMA_BF16(a1, b3, acc[1][3]);
    if (s + 1 < nSteps) {
      if (wave == 0) __builtin_amdgcn_s_wait_tensorcnt(0);
      __syncthreads();          // release next buffer; also fences this buffer's reads
    }
  }

  const int cr = lane & 15;
  const int hi = lane >> 4;
  #pragma unroll
  for (int i = 0; i < 2; ++i) {
    #pragma unroll
    for (int jj = 0; jj < 4; ++jj) {
      const long long col = nBase + wn * 64 + jj * 16 + cr;
      const float bv = bias ? bias[col] : 0.f;
      #pragma unroll
      for (int r = 0; r < 8; ++r) {
        const long long row = mBase + wm * 32 + i * 16 + hi * 8 + r;
        float v = acc[i][jj][r] + bv;
        if (relu) v = fmaxf(v, 0.f);
        const long long idx = row * N + col;
        if (Cf) Cf[idx] = v;
        if (Cb) Cb[idx] = f2bf(v);
      }
    }
  }
}
#endif // HAVE_TDM

// ---------------- direct-load WMMA GEMM (fallback; also kept compiled) ----------------
__global__ __launch_bounds__(256) void gemm_bf16_kernel(
    const unsigned short* __restrict__ A,
    const unsigned short* __restrict__ W,
    const float*          __restrict__ bias,
    float*                __restrict__ Cf,
    unsigned short*       __restrict__ Cb,
    int M, int N, int K, int relu)
{
  const int lane = threadIdx.x & 31;
  const int wave = threadIdx.x >> 5;
  const int wm   = wave >> 1;
  const int wn   = wave & 1;
  const long long mBase = (long long)blockIdx.y * 128 + wm * 32;
  const long long nBase = (long long)blockIdx.x * 64  + wn * 32;

  v8f acc00, acc01, acc10, acc11;
  for (int r = 0; r < 8; ++r) { acc00[r] = 0.f; acc01[r] = 0.f; acc10[r] = 0.f; acc11[r] = 0.f; }

  const unsigned short* Ap = A + mBase * K;
  const unsigned short* Wp = W + nBase * K;
  for (int k0 = 0; k0 < K; k0 += 32) {
    v16bf a0 = load_frag_g(Ap + k0,            K, lane);
    v16bf a1 = load_frag_g(Ap + 16LL * K + k0, K, lane);
    v16bf b0 = load_frag_g(Wp + k0,            K, lane);
    v16bf b1 = load_frag_g(Wp + 16LL * K + k0, K, lane);
    acc00 = WMMA_BF16(a0, b0, acc00);
    acc01 = WMMA_BF16(a0, b1, acc01);
    acc10 = WMMA_BF16(a1, b0, acc10);
    acc11 = WMMA_BF16(a1, b1, acc11);
  }

  const int cr = lane & 15;
  const int hi = lane >> 4;
  #pragma unroll
  for (int i = 0; i < 2; ++i) {
    #pragma unroll
    for (int jj = 0; jj < 2; ++jj) {
      v8f acc = (i == 0) ? (jj == 0 ? acc00 : acc01) : (jj == 0 ? acc10 : acc11);
      const long long col = nBase + jj * 16 + cr;
      const float bv = bias ? bias[col] : 0.f;
      #pragma unroll
      for (int r = 0; r < 8; ++r) {
        const long long row = mBase + i * 16 + hi * 8 + r;
        float v = acc[r] + bv;
        if (relu) v = fmaxf(v, 0.f);
        const long long idx = row * N + col;
        if (Cf) Cf[idx] = v;
        if (Cb) Cb[idx] = f2bf(v);
      }
    }
  }
}

// ---------------- flash attention (WMMA, online softmax) ----------------
// One wave per 16-row q tile; 8 waves per block. Causal/block-causal via shift sh
// (allowed iff (k>>sh) <= (q>>sh)); key loop trimmed to kend = q0+16.
// qkv: [B, S, 3D] bf16 ; vt: [B, H, DH, S] bf16 (V transposed) ; o: [B, S, D] bf16
__global__ __launch_bounds__(256) void flash_attn_kernel(
    const unsigned short* __restrict__ qkv,
    const unsigned short* __restrict__ vt,
    unsigned short*       __restrict__ o,
    int S, int sh)
{
  __shared__ __align__(16) unsigned short pbuf[8][16 * 32];  // per-wave P bounce
  const int lane = threadIdx.x & 31;
  const int wave = threadIdx.x >> 5;
  const int b    = blockIdx.z;
  const int h    = blockIdx.y;
  const int q0   = (blockIdx.x * 8 + wave) * 16;
  const int ld   = 3 * D_;

  const unsigned short* Qb = qkv + (long long)b * S * ld + h * DH_;
  const unsigned short* Kb = Qb + D_;
  const unsigned short* Vt = vt + (long long)(b * H_ + h) * DH_ * S;

  const v16bf qa0 = load_frag_g(Qb + (long long)q0 * ld,      ld, lane);
  const v16bf qa1 = load_frag_g(Qb + (long long)q0 * ld + 32, ld, lane);

  const int hi = lane >> 4;
  const int nc = lane & 15;
  const float scale = 0.125f;                 // 1/sqrt(64)
  const int kend = q0 + 16;                   // max allowed key + 1 over this tile

  float rmax[8], rsum[8];
  v8f accO[4];
  #pragma unroll
  for (int r = 0; r < 8; ++r) { rmax[r] = -3.0e38f; rsum[r] = 0.f; }
  #pragma unroll
  for (int t = 0; t < 4; ++t)
    for (int r = 0; r < 8; ++r) accO[t][r] = 0.f;

  int qc[8];
  #pragma unroll
  for (int r = 0; r < 8; ++r) qc[r] = (q0 + r + hi * 8) >> sh;

  for (int j = 0; j < kend; j += 32) {
    // ---- scores: two 16x16 tiles (K over DH=64 -> 2 chained WMMAs each) ----
    v8f c0, c1;
    for (int r = 0; r < 8; ++r) { c0[r] = 0.f; c1[r] = 0.f; }
    {
      v16bf kb0 = load_frag_g(Kb + (long long)j * ld,      ld, lane);
      v16bf kb1 = load_frag_g(Kb + (long long)j * ld + 32, ld, lane);
      c0 = WMMA_BF16(qa0, kb0, c0);
      c0 = WMMA_BF16(qa1, kb1, c0);
    }
    const bool t1ok = (j + 16) < kend;
    if (t1ok) {
      v16bf kb0 = load_frag_g(Kb + (long long)(j + 16) * ld,      ld, lane);
      v16bf kb1 = load_frag_g(Kb + (long long)(j + 16) * ld + 32, ld, lane);
      c1 = WMMA_BF16(qa0, kb0, c1);
      c1 = WMMA_BF16(qa1, kb1, c1);
    }

    // ---- scale + mask, per-lane row-max candidate ----
    float s0[8], s1[8], mloc[8];
    #pragma unroll
    for (int r = 0; r < 8; ++r) {
      const int k0i = j + nc;
      const int k1i = j + 16 + nc;
      float a = c0[r] * scale;
      if ((k0i >> sh) > qc[r]) a = -3.0e38f;
      float bb = -3.0e38f;
      if (t1ok) {
        bb = c1[r] * scale;
        if ((k1i >> sh) > qc[r]) bb = -3.0e38f;
      }
      s0[r] = a; s1[r] = bb;
      mloc[r] = fmaxf(a, bb);
    }
    // row reduce over the 16 lanes of each half-wave (xor<16 stays in-half)
    #pragma unroll
    for (int off = 1; off < 16; off <<= 1)
      for (int r = 0; r < 8; ++r)
        mloc[r] = fmaxf(mloc[r], __shfl_xor(mloc[r], off, 32));

    // ---- online softmax update; write P (bf16) to LDS in [row][col] form ----
    float al[8], ps[8];
    #pragma unroll
    for (int r = 0; r < 8; ++r) {
      const float mnew = fmaxf(rmax[r], mloc[r]);
      al[r] = __expf(rmax[r] - mnew);
      rmax[r] = mnew;
      const float p0 = __expf(s0[r] - mnew);
      const float p1 = __expf(s1[r] - mnew);
      ps[r] = p0 + p1;
      pbuf[wave][(r + hi * 8) * 32 + nc]      = f2bf(p0);
      pbuf[wave][(r + hi * 8) * 32 + 16 + nc] = f2bf(p1);
    }
    #pragma unroll
    for (int off = 1; off < 16; off <<= 1)
      for (int r = 0; r < 8; ++r)
        ps[r] += __shfl_xor(ps[r], off, 32);
    #pragma unroll
    for (int r = 0; r < 8; ++r) rsum[r] = rsum[r] * al[r] + ps[r];
    #pragma unroll
    for (int t = 0; t < 4; ++t)
      for (int r = 0; r < 8; ++r) accO[t][r] *= al[r];

    // same-wave LDS ordering; explicit split-counter wait
    asm volatile("s_wait_dscnt 0" ::: "memory");

    // ---- re-read P as A-fragment (16x32) from LDS ----
    FragU pf;
    {
      const unsigned short* pl = &pbuf[wave][nc * 32 + hi * 8];
      pf.u[0] = *(const u32x4*)(pl);
      pf.u[1] = *(const u32x4*)(pl + 16);
    }

    // ---- O += P(16x32) @ V(32x64), V pre-transposed so kk is contiguous ----
    #pragma unroll
    for (int t = 0; t < 4; ++t) {
      v16bf vb = load_frag_g(Vt + (long long)(t * 16) * S + j, S, lane);
      accO[t] = WMMA_BF16(pf.v, vb, accO[t]);
    }
  }

  // ---- epilogue: normalize, store bf16 at [b, q, h*64 + dh] ----
  unsigned short* ob = o + ((long long)b * S + q0) * D_ + h * DH_;
  #pragma unroll
  for (int t = 0; t < 4; ++t)
    for (int r = 0; r < 8; ++r) {
      const float v = accO[t][r] / rsum[r];
      ob[(long long)(r + hi * 8) * D_ + t * 16 + nc] = f2bf(v);
    }
}

// ---------------- elementwise kernels ----------------
__global__ void convert_f32_bf16_kernel(const float* __restrict__ s,
                                        unsigned short* __restrict__ d, long long n) {
  long long i = (long long)blockIdx.x * blockDim.x + threadIdx.x;
  const long long stride = (long long)gridDim.x * blockDim.x;
  for (; i < n; i += stride) d[i] = f2bf(s[i]);
}

__global__ void embed_kernel(const int* __restrict__ src, const float* __restrict__ emb,
                             const float* __restrict__ pos, float* __restrict__ xf,
                             unsigned short* __restrict__ xb) {
  const long long n = (long long)B_ * S_PRE * D_;
  long long i = (long long)blockIdx.x * blockDim.x + threadIdx.x;
  if (i >= n) return;
  const int d  = (int)(i % D_);
  const long long bs = i / D_;
  const int s  = (int)(bs % S_PRE);
  const int tok = src[bs];
  const float v = emb[(long long)tok * D_ + d] * 32.0f + pos[(long long)s * D_ + d];
  xf[i] = v; xb[i] = f2bf(v);
}

// vt[b][h][dh][s] = qkv[b][s][2D + h*64 + dh]
__global__ void transpose_v_kernel(const unsigned short* __restrict__ qkv,
                                   unsigned short* __restrict__ vt, int S) {
  const long long n = (long long)B_ * S * D_;
  long long i = (long long)blockIdx.x * blockDim.x + threadIdx.x;
  if (i >= n) return;
  const int d  = (int)(i % D_);
  const long long bs = i / D_;
  const int s  = (int)(bs % S);
  const int b  = (int)(bs / S);
  const int h  = d >> 6, dh = d & 63;
  vt[((long long)(b * H_ + h) * DH_ + dh) * S + s] = qkv[bs * (3LL * D_) + 2 * D_ + d];
}

// x = LN(x + y) * g + beta ; writes f32 and bf16 copies. One block (256thr) per row.
__global__ __launch_bounds__(256) void add_ln_kernel(
    const float* __restrict__ x, const float* __restrict__ y,
    const float* __restrict__ g, const float* __restrict__ beta,
    float* __restrict__ xf, unsigned short* __restrict__ xb) {
  __shared__ float s1[8], s2[8];
  const long long row = blockIdx.x;
  const int t = threadIdx.x;
  const float* xr = x + row * (long long)D_;
  const float* yr = y + row * (long long)D_;
  float v[4]; float sum = 0.f, sq = 0.f;
  #pragma unroll
  for (int i = 0; i < 4; ++i) {
    const int c = t + i * 256;
    const float z = xr[c] + yr[c];
    v[i] = z; sum += z; sq += z * z;
  }
  #pragma unroll
  for (int off = 1; off < 32; off <<= 1) {
    sum += __shfl_xor(sum, off, 32);
    sq  += __shfl_xor(sq,  off, 32);
  }
  const int wv = t >> 5, ln = t & 31;
  if (ln == 0) { s1[wv] = sum; s2[wv] = sq; }
  __syncthreads();
  if (t == 0) {
    float a = 0.f, b = 0.f;
    for (int i = 0; i < 8; ++i) { a += s1[i]; b += s2[i]; }
    s1[0] = a; s2[0] = b;
  }
  __syncthreads();
  const float mean = s1[0] * (1.f / D_);
  const float var  = s2[0] * (1.f / D_) - mean * mean;
  const float rstd = rsqrtf(var + EPS_);
  #pragma unroll
  for (int i = 0; i < 4; ++i) {
    const int c = t + i * 256;
    const float z = (v[i] - mean) * rstd * g[c] + beta[c];
    xf[row * (long long)D_ + c] = z;
    xb[row * (long long)D_ + c] = f2bf(z);
  }
}

// x2[b][sc][d] = mean over 4 chunk rows
__global__ void pool_kernel(const float* __restrict__ x, float* __restrict__ x2f,
                            unsigned short* __restrict__ x2b) {
  const long long n = (long long)B_ * S_POST * D_;
  long long i = (long long)blockIdx.x * blockDim.x + threadIdx.x;
  if (i >= n) return;
  const int d = (int)(i % D_);
  const long long bsc = i / D_;
  const int sc = (int)(bsc % S_POST);
  const int b  = (int)(bsc / S_POST);
  const float* xr = x + ((long long)b * S_PRE + sc * CF_) * D_ + d;
  const float s = 0.25f * (xr[0] + xr[D_] + xr[2 * D_] + xr[3 * D_]);
  x2f[i] = s; x2b[i] = f2bf(s);
}

// ---------------- host orchestration ----------------
extern "C" void kernel_launch(void* const* d_in, const int* in_sizes, int n_in,
                              void* d_out, int out_size, void* d_ws, size_t ws_size,
                              hipStream_t stream) {
  (void)in_sizes; (void)n_in; (void)out_size; (void)ws_size;

  const int*   src    = (const int*)  d_in[0];
  const float* emb    = (const float*)d_in[1];
  const float* pos    = (const float*)d_in[2];
  const float* lin_w  = (const float*)d_in[3];
  const float* lin_b  = (const float*)d_in[4];
  const float* pre_w[12];   // in_w,in_b,out_w,out_b,ln1w,ln1b,ln2w,ln2b,f1w,f1b,f2w,f2b
  const float* post_w[12];
  for (int i = 0; i < 12; ++i) { pre_w[i]  = (const float*)d_in[5 + i];
                                 post_w[i] = (const float*)d_in[17 + i]; }

  char* wsb = (char*)d_ws;
  size_t off = 0;
  auto alloc = [&](size_t bytes) -> void* {
    void* p = wsb + off;
    off += (bytes + 255) & ~(size_t)255;
    return p;
  };

  unsigned short* wb_pre_in  = (unsigned short*)alloc(2ULL * 3 * D_ * D_ * 2);
  unsigned short* wb_pre_out = (unsigned short*)alloc(2ULL * D_ * D_ * 2);
  unsigned short* wb_pre_f1  = (unsigned short*)alloc(2ULL * FF_ * D_ * 2);
  unsigned short* wb_pre_f2  = (unsigned short*)alloc(2ULL * D_ * FF_ * 2);
  unsigned short* wb_post_in  = (unsigned short*)alloc(2ULL * 3 * D_ * D_ * 2);
  unsigned short* wb_post_out = (unsigned short*)alloc(2ULL * D_ * D_ * 2);
  unsigned short* wb_post_f1  = (unsigned short*)alloc(2ULL * FF_ * D_ * 2);
  unsigned short* wb_post_f2  = (unsigned short*)alloc(2ULL * D_ * FF_ * 2);
  unsigned short* wb_lin      = (unsigned short*)alloc((unsigned long long)NT_ * D_ * 2);

  float*          x_f   = (float*)         alloc(4ULL * B_ * S_PRE * D_);
  unsigned short* x_b   = (unsigned short*)alloc(2ULL * B_ * S_PRE * D_);
  unsigned short* qkv_b = (unsigned short*)alloc(2ULL * B_ * S_PRE * 3 * D_);
  unsigned short* vt_b  = (unsigned short*)alloc(2ULL * B_ * S_PRE * D_ + 4096); // +pad for tail reads
  unsigned short* o_b   = (unsigned short*)alloc(2ULL * B_ * S_PRE * D_);
  float*          y_f   = (float*)         alloc(4ULL * B_ * S_PRE * D_);
  unsigned short* h_b   = (unsigned short*)alloc(2ULL * B_ * S_PRE * FF_);
  float*          x2_f  = (float*)         alloc(4ULL * B_ * S_POST * D_);
  unsigned short* x2_b  = (unsigned short*)alloc(2ULL * B_ * S_POST * D_);

  auto conv = [&](const float* s, unsigned short* d, long long n) {
    convert_f32_bf16_kernel<<<1024, 256, 0, stream>>>(s, d, n);
  };
  auto gemm = [&](const unsigned short* A, const unsigned short* W, const float* bias,
                  float* Cf, unsigned short* Cb, int M, int N, int K, int relu) {
#ifdef HAVE_TDM
    dim3 g(N / 128, M / 128);
    gemm_bf16_tdm_kernel<<<g, 256, 0, stream>>>(A, W, bias, Cf, Cb, M, N, K, relu);
#else
    dim3 g(N / 64, M / 128);
    gemm_bf16_kernel<<<g, 256, 0, stream>>>(A, W, bias, Cf, Cb, M, N, K, relu);
#endif
  };

  // ---- weight conversion (f32 -> bf16, L2-resident working set ~160MB) ----
  conv(pre_w[0],  wb_pre_in,  2LL * 3 * D_ * D_);
  conv(pre_w[2],  wb_pre_out, 2LL * D_ * D_);
  conv(pre_w[8],  wb_pre_f1,  2LL * FF_ * D_);
  conv(pre_w[10], wb_pre_f2,  2LL * D_ * FF_);
  conv(post_w[0],  wb_post_in,  2LL * 3 * D_ * D_);
  conv(post_w[2],  wb_post_out, 2LL * D_ * D_);
  conv(post_w[8],  wb_post_f1,  2LL * FF_ * D_);
  conv(post_w[10], wb_post_f2,  2LL * D_ * FF_);
  conv(lin_w, wb_lin, (long long)NT_ * D_);

  // ---- embedding ----
  {
    const long long n = (long long)B_ * S_PRE * D_;
    embed_kernel<<<(unsigned)((n + 255) / 256), 256, 0, stream>>>(src, emb, pos, x_f, x_b);
  }

  auto enc_layer = [&](float* xf, unsigned short* xb, int M, int Sl, int sh,
                       const unsigned short* w_in, const float* b_in,
                       const unsigned short* w_out, const float* b_out,
                       const float* ln1w, const float* ln1b,
                       const float* ln2w, const float* ln2b,
                       const unsigned short* w_f1, const float* b_f1,
                       const unsigned short* w_f2, const float* b_f2) {
    gemm(xb, w_in, b_in, nullptr, qkv_b, M, 3 * D_, D_, 0);
    const long long n = (long long)B_ * Sl * D_;
    transpose_v_kernel<<<(unsigned)((n + 255) / 256), 256, 0, stream>>>(qkv_b, vt_b, Sl);
    dim3 ag(Sl / 128, H_, B_);
    flash_attn_kernel<<<ag, 256, 0, stream>>>(qkv_b, vt_b, o_b, Sl, sh);
    gemm(o_b, w_out, b_out, y_f, nullptr, M, D_, D_, 0);
    add_ln_kernel<<<M, 256, 0, stream>>>(xf, y_f, ln1w, ln1b, xf, xb);
    gemm(xb, w_f1, b_f1, nullptr, h_b, M, FF_, D_, 1);
    gemm(h_b, w_f2, b_f2, y_f, nullptr, M, D_, FF_, 0);
    add_ln_kernel<<<M, 256, 0, stream>>>(xf, y_f, ln2w, ln2b, xf, xb);
  };

  // ---- pre layers (S=2048, block-causal chunk=4 -> shift 2) ----
  for (int i = 0; i < 2; ++i)
    enc_layer(x_f, x_b, B_ * S_PRE, S_PRE, 2,
              wb_pre_in  + (long long)i * 3 * D_ * D_, pre_w[1] + i * 3 * D_,
              wb_pre_out + (long long)i * D_ * D_,     pre_w[3] + i * D_,
              pre_w[4] + i * D_,  pre_w[5] + i * D_,
              pre_w[6] + i * D_,  pre_w[7] + i * D_,
              wb_pre_f1 + (long long)i * FF_ * D_,     pre_w[9]  + i * FF_,
              wb_pre_f2 + (long long)i * D_ * FF_,     pre_w[11] + i * D_);

  // ---- chunk average pool ----
  {
    const long long n = (long long)B_ * S_POST * D_;
    pool_kernel<<<(unsigned)((n + 255) / 256), 256, 0, stream>>>(x_f, x2_f, x2_b);
  }

  // ---- post layers (S=512, causal -> shift 0) ----
  for (int i = 0; i < 2; ++i)
    enc_layer(x2_f, x2_b, B_ * S_POST, S_POST, 0,
              wb_post_in  + (long long)i * 3 * D_ * D_, post_w[1] + i * 3 * D_,
              wb_post_out + (long long)i * D_ * D_,     post_w[3] + i * D_,
              post_w[4] + i * D_,  post_w[5] + i * D_,
              post_w[6] + i * D_,  post_w[7] + i * D_,
              wb_post_f1 + (long long)i * FF_ * D_,     post_w[9]  + i * FF_,
              wb_post_f2 + (long long)i * D_ * FF_,     post_w[11] + i * D_);

  // ---- vocab projection straight into d_out: [B*S_POST, NT] f32 ----
  gemm(x2_b, wb_lin, lin_b, (float*)d_out, nullptr, B_ * S_POST, NT_, D_, 0);
}
